// RNN_62672162783550
// MI455X (gfx1250) — 8-bit matrix operand order repaired and checked by reading, compiled, not run on hardware
//
#include <hip/hip_runtime.h>
#include <hip/hip_bf16.h>

#define VOCAB 400000
#define EMB_D 50
#define RNN_D 512
#define HID_D 1024
#define SEQ   8192

typedef __attribute__((ext_vector_type(16))) int            v16i;
typedef __attribute__((ext_vector_type(8)))  float          v8f;
typedef __attribute__((ext_vector_type(16))) unsigned short v16u16;
typedef __attribute__((ext_vector_type(16))) __bf16         v16bf;
typedef __attribute__((ext_vector_type(4)))  unsigned int   v4u;
typedef __attribute__((ext_vector_type(8)))  int            v8i_;
typedef __attribute__((ext_vector_type(4)))  int            v4i_;

#if defined(__has_builtin)
#if __has_builtin(__builtin_amdgcn_tensor_load_to_lds) && __has_builtin(__builtin_amdgcn_s_wait_tensorcnt)
#define USE_TDM 1
#endif
#endif
#ifndef USE_TDM
#define USE_TDM 0
#endif
#if __has_include(<hip/amd_detail/amd_gfx1250_TDM.h>)
#define TDM_SIX_ARGS 1
#else
#define TDM_SIX_ARGS 0
#endif

#if USE_TDM
__device__ __forceinline__ void tdm_issue(v4u g0, v8i_ g1, v4i_ g2, v4i_ g3) {
#if TDM_SIX_ARGS
  v8i_ z = {0, 0, 0, 0, 0, 0, 0, 0};
  __builtin_amdgcn_tensor_load_to_lds(g0, g1, g2, g3, z, 0);
#else
  __builtin_amdgcn_tensor_load_to_lds(g0, g1, g2, g3, 0);
#endif
}
#endif


__device__ __forceinline__ unsigned short f32_to_bf16_bits(float f) {
  union { float f; unsigned u; } v; v.f = f;
  unsigned u = v.u;
  u += 0x7fffu + ((u >> 16) & 1u);
  return (unsigned short)(u >> 16);
}

__device__ __forceinline__ unsigned char f32_to_fp8_e4m3(float x) {
  union { float f; unsigned u; } v; v.f = x;
  unsigned sign = (v.u >> 31) << 7;
  float ax = fabsf(x);
  if (!(ax >= 0x1p-9f)) return (unsigned char)sign;
  if (ax > 448.0f) ax = 448.0f;
  v.f = ax;
  int e = (int)((v.u >> 23) & 0xffu) - 127;
  if (e < -6) {
    int mden = (int)(ax * 512.0f + 0.5f);
    if (mden >= 8) return (unsigned char)(sign | 0x08u);
    return (unsigned char)(sign | (unsigned)mden);
  }
  unsigned mant = (v.u >> 20) & 0x7u;
  unsigned rest = v.u & 0xfffffu;
  if (rest > 0x80000u || (rest == 0x80000u && (mant & 1u))) {
    mant++;
    if (mant == 8u) { mant = 0u; e++; }
  }
  if (e > 8 || (e == 8 && mant > 6u)) { e = 8; mant = 6u; }
  return (unsigned char)(sign | ((unsigned)(e + 7) << 3) | mant);
}


__global__ void prep_wih_kernel(const float* __restrict__ W_ih,
                                unsigned short* __restrict__ frag) {
  int i = blockIdx.x * blockDim.x + threadIdx.x;
  if (i >= 32 * 2 * 32 * 16) return;
  int e    = i & 15;
  int lane = (i >> 4) & 31;
  int kc   = (i >> 9) & 1;
  int nt   = i >> 10;
  int n = nt * 16 + (lane & 15);
  int k = kc * 32 + (e < 8 ? e : e + 8) + ((lane & 16) ? 8 : 0);
  float val = (k < EMB_D) ? W_ih[n * EMB_D + k] : 0.0f;
  frag[i] = f32_to_bf16_bits(val);
}


__global__ void prep_whh_kernel(const float* __restrict__ W_hh,
                                unsigned int* __restrict__ frag) {
  int i = blockIdx.x * blockDim.x + threadIdx.x;
  if (i >= 32 * 4 * 32 * 16) return;
  int d    = i & 15;
  int lane = (i >> 4) & 31;
  int kc   = (i >> 9) & 3;
  int nt   = i >> 11;
  int n = nt * 16 + (lane & 15);
  int kbase = kc * 128 + 32 * (d >> 2) + ((lane & 16) ? 16 : 0) + (d & 3) * 4;
  unsigned int w = 0;
  #pragma unroll
  for (int b = 0; b < 4; ++b)
    w |= ((unsigned int)f32_to_fp8_e4m3(W_hh[(size_t)n * RNN_D + kbase + b])) << (8 * b);
  frag[i] = w;
}


__global__ __launch_bounds__(256)
void embed_project_kernel(const int* __restrict__ s1, const int* __restrict__ s2,
                          const float* __restrict__ emb,
                          const unsigned short* __restrict__ wih_frag,
                          float* __restrict__ xp) {
  __shared__ float          stage[16][EMB_D];
  __shared__ unsigned short atile[16][64];
  int bid = blockIdx.x;
  int s   = bid & 1;
  int mt  = bid >> 1;
  const int* seq = s ? s2 : s1;
  int tid = threadIdx.x;

#if USE_TDM
  if (tid < 32) {
    unsigned lds0  = (unsigned)(size_t)&stage[0][0];
    unsigned long long ga = (unsigned long long)(size_t)emb;
    unsigned g0w3 = (unsigned)((ga >> 32) & 0x01ffffffu) | 0x80000000u;
    v4u g0a = { 0xC0000001u, lds0,                  (unsigned)(ga & 0xffffffffu), g0w3 };
    v4u g0b = { 0xC0000001u, lds0 + 8u * EMB_D * 4u, (unsigned)(ga & 0xffffffffu), g0w3 };
    v8i_ g1 = { (int)(2u << 16),
                (int)((unsigned)EMB_D << 16),
                (int)(((unsigned)VOCAB & 0xffffu) << 16),
                (int)(((unsigned)VOCAB >> 16) | ((unsigned)EMB_D << 16)),
                8, EMB_D, 0, 0 };
    int tb = mt * 16;
    v4i_ i0 = { seq[tb + 0], seq[tb + 1], seq[tb + 2],  seq[tb + 3]  };
    v4i_ i1 = { seq[tb + 4], seq[tb + 5], seq[tb + 6],  seq[tb + 7]  };
    v4i_ i2 = { seq[tb + 8], seq[tb + 9], seq[tb + 10], seq[tb + 11] };
    v4i_ i3 = { seq[tb + 12], seq[tb + 13], seq[tb + 14], seq[tb + 15] };
    tdm_issue(g0a, g1, i0, i1);
    tdm_issue(g0b, g1, i2, i3);
    __builtin_amdgcn_s_wait_tensorcnt(0);
  }
#else
  for (int e = tid; e < 16 * EMB_D; e += 256) {
    int r = e / EMB_D, k = e - r * EMB_D;
    stage[r][k] = emb[(size_t)seq[mt * 16 + r] * EMB_D + k];
  }
#endif
  __syncthreads();
  for (int e = tid; e < 16 * 64; e += 256) {
    int r = e >> 6, k = e & 63;
    atile[r][k] = f32_to_bf16_bits(k < EMB_D ? stage[r][k] : 0.0f);
  }
  __syncthreads();

  int lane = tid & 31, wave = tid >> 5;
  int m = lane & 15;
  int khi = (lane & 16) ? 8 : 0;
  v16u16 au0, au1;
  #pragma unroll
  for (int e = 0; e < 16; ++e) {
    int k = (e < 8 ? e : e + 8) + khi;
    au0[e] = atile[m][k];
    au1[e] = atile[m][32 + k];
  }
  v16bf a0 = __builtin_bit_cast(v16bf, au0);
  v16bf a1 = __builtin_bit_cast(v16bf, au1);

  float* xps = xp + (size_t)s * SEQ * RNN_D;
  int rowoff = (lane & 16) ? 8 : 0;
  for (int j = 0; j < 4; ++j) {
    int nt = wave * 4 + j;
    v16u16 bu0 = *(const v16u16*)(wih_frag + ((size_t)((nt * 2 + 0) * 32 + lane)) * 16);
    v16u16 bu1 = *(const v16u16*)(wih_frag + ((size_t)((nt * 2 + 1) * 32 + lane)) * 16);
    v16bf b0 = __builtin_bit_cast(v16bf, bu0);
    v16bf b1 = __builtin_bit_cast(v16bf, bu1);
    v8f c = {};
    c = __builtin_amdgcn_wmma_f32_16x16x32_bf16(false, a0, false, b0, (short)0, c, false, false);
    c = __builtin_amdgcn_wmma_f32_16x16x32_bf16(false, a1, false, b1, (short)0, c, false, false);
    int col = nt * 16 + m;
    #pragma unroll
    for (int r = 0; r < 8; ++r)
      xps[(size_t)(mt * 16 + r + rowoff) * RNN_D + col] = c[r];
  }
}


__global__ __launch_bounds__(512, 1)
void rnn_recur_kernel(const float* __restrict__ xp,
                      const unsigned int* __restrict__ whh_frag,
                      float* __restrict__ hT) {
  __shared__ unsigned char h8[2][2][RNN_D];
  int tid = threadIdx.x, lane = tid & 31, wave = tid >> 5;
  ((unsigned int*)h8)[tid] = 0u;

  int m = lane & 15;
  int khi = (lane & 16) ? 8 : 0;
  int nt0 = wave * 2, nt1 = wave * 2 + 1;

  v16i B0[4], B1[4];
  #pragma unroll
  for (int kc = 0; kc < 4; ++kc) {
    B0[kc] = *(const v16i*)(whh_frag + ((size_t)((nt0 * 4 + kc) * 32 + lane)) * 16);
    B1[kc] = *(const v16i*)(whh_frag + ((size_t)((nt1 * 4 + kc) * 32 + lane)) * 16);
  }

  int koff[16];
  #pragma unroll
  for (int d = 0; d < 16; ++d)
    koff[d] = (32 * (d >> 2) + (d & 3) * 4 + 2 * khi) >> 2;

  const float* xp1 = xp;
  const float* xp2 = xp + (size_t)SEQ * RNN_D;
  int n0 = nt0 * 16 + m;
  int n1 = nt1 * 16 + m;
  __syncthreads();

  for (int t = 0; t < SEQ; ++t) {
    int p = t & 1;
    size_t base = (size_t)t * RNN_D;
    float x10 = xp1[base + n0];
    float x20 = xp2[base + n0];
    float x11 = xp1[base + n1];
    float x21 = xp2[base + n1];
    if (t + 8 < SEQ) {
      __builtin_prefetch(xp1 + base + 8 * RNN_D + n0, 0, 1);
      __builtin_prefetch(xp2 + base + 8 * RNN_D + n0, 0, 1);
    }
    const unsigned int* hb = (const unsigned int*)&h8[p][m & 1][0];
    v8f c0 = {}, c1 = {};
    #pragma unroll
    for (int kc = 0; kc < 4; ++kc) {
      v16i a;
      #pragma unroll
      for (int d = 0; d < 16; ++d)
        a[d] = (m < 2) ? (int)hb[kc * 32 + koff[d]] : 0;
      c0 = __builtin_amdgcn_wmma_f32_16x16x128_fp8_fp8(a, B0[kc], (short)0, c0, false, false);
      c1 = __builtin_amdgcn_wmma_f32_16x16x128_fp8_fp8(a, B1[kc], (short)0, c1, false, false);
    }
    if (lane < 16) {
      float nh10 = tanhf(c0[0] + x10);
      float nh20 = tanhf(c0[1] + x20);
      float nh11 = tanhf(c1[0] + x11);
      float nh21 = tanhf(c1[1] + x21);
      int q = p ^ 1;
      h8[q][0][n0] = f32_to_fp8_e4m3(nh10);
      h8[q][1][n0] = f32_to_fp8_e4m3(nh20);
      h8[q][0][n1] = f32_to_fp8_e4m3(nh11);
      h8[q][1][n1] = f32_to_fp8_e4m3(nh21);
      if (t == SEQ - 1) {
        hT[n0] = nh10; hT[RNN_D + n0] = nh20;
        hT[n1] = nh11; hT[RNN_D + n1] = nh21;
      }
    }
    __syncthreads();
  }
}


__global__ __launch_bounds__(256)
void mlp_kernel(const float* __restrict__ hT, const float* __restrict__ W_hid,
                const float* __restrict__ b_hid, const float* __restrict__ W_out,
                const float* __restrict__ b_out, float* __restrict__ out) {
  __shared__ float cat[2 * RNN_D];
  __shared__ float red[256];
  int tid = threadIdx.x;
  for (int i = tid; i < 2 * RNN_D; i += 256) cat[i] = hT[i];
  __syncthreads();
  float acc = 0.0f;
  for (int i = tid; i < HID_D; i += 256) {
    float s = b_hid[i];
    const float* wr = W_hid + (size_t)i * (2 * RNN_D);
    for (int j = 0; j < 2 * RNN_D; ++j) s = fmaf(wr[j], cat[j], s);
    acc = fmaf(W_out[i], fmaxf(s, 0.0f), acc);
  }
  red[tid] = acc;
  __syncthreads();
  for (int off = 128; off > 0; off >>= 1) {
    if (tid < off) red[tid] += red[tid + off];
    __syncthreads();
  }
  if (tid == 0) out[0] = 1.0f / (1.0f + expf(-(red[0] + b_out[0])));
}


extern "C" void kernel_launch(void* const* d_in, const int* in_sizes, int n_in,
                              void* d_out, int out_size, void* d_ws, size_t ws_size,
                              hipStream_t stream) {
  const int*   s1    = (const int*)d_in[0];
  const int*   s2    = (const int*)d_in[1];
  const float* emb   = (const float*)d_in[2];
  const float* W_ih  = (const float*)d_in[3];
  const float* W_hh  = (const float*)d_in[4];
  const float* W_hid = (const float*)d_in[5];
  const float* b_hid = (const float*)d_in[6];
  const float* W_out = (const float*)d_in[7];
  const float* b_out = (const float*)d_in[8];
  (void)in_sizes; (void)n_in; (void)out_size; (void)ws_size;

  char* ws = (char*)d_ws;
  float*          ws_xp  = (float*)ws;
  unsigned short* ws_wih = (unsigned short*)(ws + 33554432);
  unsigned int*   ws_whh = (unsigned int*)(ws + 33554432 + 65536);
  float*          ws_hT  = (float*)(ws + 33554432 + 65536 + 262144);

  prep_wih_kernel<<<128, 256, 0, stream>>>(W_ih, ws_wih);
  prep_whh_kernel<<<256, 256, 0, stream>>>(W_hh, ws_whh);
  embed_project_kernel<<<1024, 256, 0, stream>>>(s1, s2, emb, ws_wih, ws_xp);
  rnn_recur_kernel<<<1, 512, 0, stream>>>(ws_xp, ws_whh, ws_hT);
  mlp_kernel<<<1, 256, 0, stream>>>(ws_hT, W_hid, b_hid, W_out, b_out, (float*)d_out);
}
